// MHTVI_38543036514605
// MI455X (gfx1250) — compile-verified, matrix operand order checked by reading, not run on hardware
//
#include <hip/hip_runtime.h>
#include <hip/hip_bf16.h>

typedef __attribute__((ext_vector_type(16))) __bf16 v16bf;
typedef __attribute__((ext_vector_type(8)))  float  v8f;

// B=32 ENC=512 SEQ=1024 H=8 HD=128 ; M = B*ENC = 16384
// scale = 1/sqrt(HD*ENC) = 1/256

// ---------------------------------------------------------------------------
// WMMA helpers (CDNA5 layouts per ISA 7.12.2)
// ---------------------------------------------------------------------------
__device__ __forceinline__ v8f wmma_bf16(v16bf a, v16bf b, v8f c) {
  return __builtin_amdgcn_wmma_f32_16x16x32_bf16(false, a, false, b,
                                                 (short)0, c, false, false);
}

// Load a 16x32 bf16 fragment (A-layout; also used for B = W^T since the
// per-lane pattern is identical with lane -> output column, row of W).
// lane L: row = row_base + (L & 15); K chunks at k_base+(L>>4)*8 and +16.
__device__ __forceinline__ v16bf load_frag_bf16(const __hip_bfloat16* base,
                                                int ld, int row_base,
                                                int k_base, int lane) {
  union { v16bf v; uint4 q[2]; } u;
  const uint4* p = (const uint4*)(base +
      (size_t)(row_base + (lane & 15)) * ld + (k_base + ((lane >> 4) << 3)));
  u.q[0] = p[0];
  u.q[1] = p[2];   // +16 elements = +32 bytes
  return u.v;
}

// ---------------------------------------------------------------------------
// f32 -> bf16 weight conversion
// ---------------------------------------------------------------------------
__global__ void mhtvi_cvt_kernel(const float* __restrict__ src,
                                 __hip_bfloat16* __restrict__ dst, int n) {
  int i = blockIdx.x * blockDim.x + threadIdx.x;
  int stride = gridDim.x * blockDim.x;
  for (; i < n; i += stride) dst[i] = __float2bfloat16(src[i]);
}

// ---------------------------------------------------------------------------
// LayerNorm over last dim (1024) + bf16 output.  One block (256 thr) per row.
// ---------------------------------------------------------------------------
__global__ void mhtvi_ln_kernel(const float* __restrict__ x,
                                const float* __restrict__ g,
                                const float* __restrict__ bta,
                                __hip_bfloat16* __restrict__ xn) {
  __shared__ float red[16];
  const int row = blockIdx.x;
  const int t   = threadIdx.x;
  const float* xr = x + (size_t)row * 1024;
  float v[4]; float sum = 0.f, sq = 0.f;
#pragma unroll
  for (int i = 0; i < 4; ++i) {
    v[i] = xr[t + (i << 8)];
    sum += v[i]; sq += v[i] * v[i];
  }
#pragma unroll
  for (int off = 16; off > 0; off >>= 1) {
    sum += __shfl_xor(sum, off);
    sq  += __shfl_xor(sq, off);
  }
  if ((t & 31) == 0) { red[t >> 5] = sum; red[8 + (t >> 5)] = sq; }
  __syncthreads();
  float ts = 0.f, tq = 0.f;
#pragma unroll
  for (int w = 0; w < 8; ++w) { ts += red[w]; tq += red[8 + w]; }
  const float mu  = ts * (1.0f / 1024.0f);
  const float var = tq * (1.0f / 1024.0f) - mu * mu;
  const float inv = rsqrtf(var + 1e-5f);
#pragma unroll
  for (int i = 0; i < 4; ++i) {
    const int s = t + (i << 8);
    const float y = (v[i] - mu) * inv * g[s] + bta[s];
    xn[(size_t)row * 1024 + s] = __float2bfloat16(y);
  }
}

// ---------------------------------------------------------------------------
// GEMM1: qkv = xn[16384x1024] @ w_in^T[1024x3072] + b_in, scatter to the
// reshaped [B,H,512,128] q/k/v buffers (bf16).
// One wave computes a 32x64 tile: 2 A-frags x 4 B-frags -> 8 WMMAs / k-step.
// ---------------------------------------------------------------------------
__global__ void mhtvi_gemm_qkv_kernel(const __hip_bfloat16* __restrict__ xn,
                                      const __hip_bfloat16* __restrict__ w,
                                      const float* __restrict__ b_in,
                                      __hip_bfloat16* __restrict__ q_r,
                                      __hip_bfloat16* __restrict__ k_r,
                                      __hip_bfloat16* __restrict__ v_r) {
  const int lane  = threadIdx.x & 31;
  const int wid   = (blockIdx.x * blockDim.x + threadIdx.x) >> 5;
  const int mBase = (wid & 511) << 5;    // 512 tiles of 32 rows
  const int nBase = (wid >> 9) << 6;     // 48 groups of 64 cols
  union { v8f v; float e[8]; } c[2][4];
#pragma unroll
  for (int r = 0; r < 2; ++r)
#pragma unroll
    for (int f = 0; f < 4; ++f) c[r][f].v = (v8f){0,0,0,0,0,0,0,0};
  for (int k = 0; k < 1024; k += 32) {
    if (k + 128 < 1024) {   // pull next A k-slices toward the WGP
      __builtin_prefetch((const char*)(xn +
          (size_t)(mBase + (lane & 15)) * 1024 + (k + 128)), 0, 1);
      __builtin_prefetch((const char*)(xn +
          (size_t)(mBase + 16 + (lane & 15)) * 1024 + (k + 128)), 0, 1);
    }
    const v16bf a0 = load_frag_bf16(xn, 1024, mBase,      k, lane);
    const v16bf a1 = load_frag_bf16(xn, 1024, mBase + 16, k, lane);
#pragma unroll
    for (int f = 0; f < 4; ++f) {
      const v16bf b = load_frag_bf16(w, 1024, nBase + (f << 4), k, lane);
      c[0][f].v = wmma_bf16(a0, b, c[0][f].v);
      c[1][f].v = wmma_bf16(a1, b, c[1][f].v);
    }
  }
  const int nCol = lane & 15;
  const int mOff = (lane >> 4) << 3;
#pragma unroll
  for (int f = 0; f < 4; ++f) {
    const int j = nBase + (f << 4) + nCol;
    __hip_bfloat16* dst; int s;
    if (j < 1024)      { dst = q_r; s = j; }
    else if (j < 2048) { dst = k_r; s = j - 1024; }
    else               { dst = v_r; s = j - 2048; }
    const float bias = b_in[j];
    const int d = s & 127, shi = s >> 7;
#pragma unroll
    for (int r = 0; r < 2; ++r) {
#pragma unroll
      for (int i = 0; i < 8; ++i) {
        const int m  = mBase + (r << 4) + mOff + i;
        const int b_ = m >> 9, n = m & 511;
        const int h  = n >> 6;
        const int n2 = ((n & 63) << 3) | shi;
        const int idx = (((b_ << 3) | h) << 16) | (n2 << 7) | d;
        dst[idx] = __float2bfloat16(c[r][f].e[i] + bias);
      }
    }
  }
}

// ---------------------------------------------------------------------------
// mlp1: h1 = relu(T[131072x128] @ w_m1a^T + b_m1a); out = h1 @ w_m1b + b_m1b
// One wave per 16-row tile (K=128, full N=128 -> 8 C frags, 32 wmma/wave).
// ---------------------------------------------------------------------------
__global__ void mhtvi_mlp1_kernel(const __hip_bfloat16* __restrict__ T,
                                  const __hip_bfloat16* __restrict__ wa,
                                  const float* __restrict__ b1a,
                                  const float* __restrict__ w1b,
                                  const float* __restrict__ b1b,
                                  float* __restrict__ outv) {
  const int lane  = threadIdx.x & 31;
  const int wid   = (blockIdx.x * blockDim.x + threadIdx.x) >> 5;
  const int mBase = wid << 4;            // 8192 tiles
  union { v8f v; float e[8]; } c[8];
#pragma unroll
  for (int f = 0; f < 8; ++f) c[f].v = (v8f){0,0,0,0,0,0,0,0};
#pragma unroll
  for (int k = 0; k < 128; k += 32) {
    const v16bf a = load_frag_bf16(T, 128, mBase, k, lane);
#pragma unroll
    for (int f = 0; f < 8; ++f) {
      const v16bf b = load_frag_bf16(wa, 128, f << 4, k, lane);
      c[f].v = wmma_bf16(a, b, c[f].v);
    }
  }
  const int nCol = lane & 15;
  float s[8];
#pragma unroll
  for (int i = 0; i < 8; ++i) s[i] = 0.f;
#pragma unroll
  for (int f = 0; f < 8; ++f) {
    const int col = (f << 4) + nCol;
    const float bias = b1a[col], wc = w1b[col];
#pragma unroll
    for (int i = 0; i < 8; ++i) {
      float h1 = c[f].e[i] + bias;
      h1 = h1 > 0.f ? h1 : 0.f;
      s[i] += h1 * wc;
    }
  }
  // reduce across the 16 lanes holding one row (xor<16 stays in-group)
#pragma unroll
  for (int i = 0; i < 8; ++i) {
    s[i] += __shfl_xor(s[i], 1);
    s[i] += __shfl_xor(s[i], 2);
    s[i] += __shfl_xor(s[i], 4);
    s[i] += __shfl_xor(s[i], 8);
  }
  if (nCol == 0) {
    const int rowBase = mBase + ((lane >> 4) << 3);
    const float bb = b1b[0];
#pragma unroll
    for (int i = 0; i < 8; ++i) outv[rowBase + i] = s[i] + bb;
  }
}

// ---------------------------------------------------------------------------
// mlp2: out[bh,d] = relu(sum_n T[bh,n,d]*w2[n] + b2). One block per (b,h).
// ---------------------------------------------------------------------------
__global__ void mhtvi_mlp2_kernel(const __hip_bfloat16* __restrict__ T,
                                  const float* __restrict__ w2,
                                  const float* __restrict__ b2,
                                  float* __restrict__ outv) {
  const int bh = blockIdx.x;      // 256
  const int d  = threadIdx.x;     // 128
  const __hip_bfloat16* base = T + ((size_t)bh << 16) + d;
  float acc = 0.f;
  for (int n = 0; n < 512; ++n)
    acc += __bfloat162float(base[n << 7]) * w2[n];
  acc += b2[0];
  outv[(bh << 7) + d] = acc > 0.f ? acc : 0.f;
}

// ---------------------------------------------------------------------------
// combine: v1/v2 = mlp1(.)[n2] * mlp2(.)[d] * scale * v ; un-reshape and
// write the concat buffer [16384 x 2048] bf16.
// ---------------------------------------------------------------------------
__global__ void mhtvi_combine_kernel(const __hip_bfloat16* __restrict__ v_r,
                                     const float* __restrict__ a_q,
                                     const float* __restrict__ a_k,
                                     const float* __restrict__ c_q,
                                     const float* __restrict__ c_k,
                                     __hip_bfloat16* __restrict__ cat) {
  const int idx = blockIdx.x * blockDim.x + threadIdx.x;  // < 16777216
  const int d  = idx & 127;
  const int n2 = (idx >> 7) & 511;
  const int bh = idx >> 16;
  const float vv = __bfloat162float(v_r[idx]);
  const float scale = 1.0f / 256.0f;      // 1/sqrt(128*512)
  const float v1 = a_q[(bh << 9) | n2] * c_k[(bh << 7) | d] * scale * vv;
  const float v2 = a_k[(bh << 9) | n2] * c_q[(bh << 7) | d] * scale * vv;
  const int h = bh & 7, b_ = bh >> 3;
  const int n = (h << 6) | (n2 >> 3);
  const int s = ((n2 & 7) << 7) | d;
  const size_t row = (size_t)((b_ << 9) | n);
  cat[row * 2048 + s]        = __float2bfloat16(v1);
  cat[row * 2048 + 1024 + s] = __float2bfloat16(v2);
}

// ---------------------------------------------------------------------------
// GEMM2: out = cat[16384x2048] @ w_out^T[2048x1024] + b_out + x  (f32 out)
// One wave computes a 32x64 tile (2 A-frags x 4 B-frags / k-step).
// ---------------------------------------------------------------------------
__global__ void mhtvi_gemm_out_kernel(const __hip_bfloat16* __restrict__ cat,
                                      const __hip_bfloat16* __restrict__ w,
                                      const float* __restrict__ b_out,
                                      const float* __restrict__ x,
                                      float* __restrict__ out) {
  const int lane  = threadIdx.x & 31;
  const int wid   = (blockIdx.x * blockDim.x + threadIdx.x) >> 5;
  const int mBase = (wid & 511) << 5;    // 512 tiles of 32 rows
  const int nBase = (wid >> 9) << 6;     // 16 groups of 64 cols
  union { v8f v; float e[8]; } c[2][4];
#pragma unroll
  for (int r = 0; r < 2; ++r)
#pragma unroll
    for (int f = 0; f < 4; ++f) c[r][f].v = (v8f){0,0,0,0,0,0,0,0};
  for (int k = 0; k < 2048; k += 32) {
    if (k + 128 < 2048) {
      __builtin_prefetch((const char*)(cat +
          (size_t)(mBase + (lane & 15)) * 2048 + (k + 128)), 0, 1);
      __builtin_prefetch((const char*)(cat +
          (size_t)(mBase + 16 + (lane & 15)) * 2048 + (k + 128)), 0, 1);
    }
    const v16bf a0 = load_frag_bf16(cat, 2048, mBase,      k, lane);
    const v16bf a1 = load_frag_bf16(cat, 2048, mBase + 16, k, lane);
#pragma unroll
    for (int f = 0; f < 4; ++f) {
      const v16bf b = load_frag_bf16(w, 2048, nBase + (f << 4), k, lane);
      c[0][f].v = wmma_bf16(a0, b, c[0][f].v);
      c[1][f].v = wmma_bf16(a1, b, c[1][f].v);
    }
  }
  const int nCol = lane & 15;
  const int mOff = (lane >> 4) << 3;
#pragma unroll
  for (int f = 0; f < 4; ++f) {
    const int j = nBase + (f << 4) + nCol;
    const float bias = b_out[j];
#pragma unroll
    for (int r = 0; r < 2; ++r) {
#pragma unroll
      for (int i = 0; i < 8; ++i) {
        const int m = mBase + (r << 4) + mOff + i;
        const size_t o = ((size_t)m << 10) + j;
        out[o] = c[r][f].e[i] + bias + x[o];
      }
    }
  }
}

// ---------------------------------------------------------------------------
// host launcher
// ---------------------------------------------------------------------------
extern "C" void kernel_launch(void* const* d_in, const int* in_sizes, int n_in,
                              void* d_out, int out_size, void* d_ws, size_t ws_size,
                              hipStream_t stream) {
  const float* x     = (const float*)d_in[0];
  const float* ln_g  = (const float*)d_in[1];
  const float* ln_b  = (const float*)d_in[2];
  const float* w_in  = (const float*)d_in[3];
  const float* b_in  = (const float*)d_in[4];
  const float* w_m1a = (const float*)d_in[5];
  const float* b_m1a = (const float*)d_in[6];
  const float* w_m1b = (const float*)d_in[7];
  const float* b_m1b = (const float*)d_in[8];
  const float* w_m2  = (const float*)d_in[9];
  const float* b_m2  = (const float*)d_in[10];
  const float* w_out = (const float*)d_in[11];
  const float* b_out = (const float*)d_in[12];
  float* out = (float*)d_out;

  char* ws = (char*)d_ws;
  // workspace layout (bytes)
  __hip_bfloat16* xn_bf    = (__hip_bfloat16*)(ws);                    // 33,554,432
  __hip_bfloat16* w_in_bf  = (__hip_bfloat16*)(ws + 33554432);         //  6,291,456
  __hip_bfloat16* w_out_bf = (__hip_bfloat16*)(ws + 39845888);         //  4,194,304
  __hip_bfloat16* w_m1a_bf = (__hip_bfloat16*)(ws + 44040192);         //     32,768
  __hip_bfloat16* q_r      = (__hip_bfloat16*)(ws + 44072960);         // 33,554,432
  __hip_bfloat16* k_r      = (__hip_bfloat16*)(ws + 77627392);         // 33,554,432
  __hip_bfloat16* v_r      = (__hip_bfloat16*)(ws + 111181824);        // 33,554,432
  float* mlp1q             = (float*)(ws + 144736256);                 //    524,288
  float* mlp1k             = (float*)(ws + 145260544);                 //    524,288
  float* mlp2q             = (float*)(ws + 145784832);                 //    131,072
  float* mlp2k             = (float*)(ws + 145915904);                 //    131,072
  // cat (64 MB) aliases q_r + k_r, which are dead once mlp1/mlp2 finish
  __hip_bfloat16* cat      = q_r;

  // 1) weights -> bf16
  mhtvi_cvt_kernel<<<(3145728 + 255) / 256, 256, 0, stream>>>(w_in,  w_in_bf,  3145728);
  mhtvi_cvt_kernel<<<(2097152 + 255) / 256, 256, 0, stream>>>(w_out, w_out_bf, 2097152);
  mhtvi_cvt_kernel<<<(16384   + 255) / 256, 256, 0, stream>>>(w_m1a, w_m1a_bf, 16384);

  // 2) layernorm -> bf16 xn
  mhtvi_ln_kernel<<<16384, 256, 0, stream>>>(x, ln_g, ln_b, xn_bf);

  // 3) qkv GEMM (512 M32-tiles x 48 N64-groups = 24576 waves)
  mhtvi_gemm_qkv_kernel<<<3072, 256, 0, stream>>>(xn_bf, w_in_bf, b_in,
                                                  q_r, k_r, v_r);

  // 4) mlp1(q), mlp1(k)   (8192 waves each)
  mhtvi_mlp1_kernel<<<1024, 256, 0, stream>>>(q_r, w_m1a_bf, b_m1a, w_m1b, b_m1b, mlp1q);
  mhtvi_mlp1_kernel<<<1024, 256, 0, stream>>>(k_r, w_m1a_bf, b_m1a, w_m1b, b_m1b, mlp1k);

  // 5) mlp2(k), mlp2(q)
  mhtvi_mlp2_kernel<<<256, 128, 0, stream>>>(k_r, w_m2, b_m2, mlp2k);
  mhtvi_mlp2_kernel<<<256, 128, 0, stream>>>(q_r, w_m2, b_m2, mlp2q);

  // 6) combine into concat buffer (aliases q_r/k_r)
  mhtvi_combine_kernel<<<65536, 256, 0, stream>>>(v_r, mlp1q, mlp1k,
                                                  mlp2q, mlp2k, cat);

  // 7) output GEMM + bias + residual (512 x 16 = 8192 waves)
  mhtvi_gemm_out_kernel<<<1024, 256, 0, stream>>>(cat, w_out_bf, b_out, x, out);
}
